// EGeoGNNModel_30365418782725
// MI455X (gfx1250) — compile-verified, matrix operand order checked by reading, not run on hardware
//
#include <hip/hip_runtime.h>
#include <math.h>

#define DD 128   // feature dim
#define HH 256   // hidden dim

typedef __bf16 bf16;
typedef __attribute__((ext_vector_type(16))) __bf16 bf16x16;
typedef __attribute__((ext_vector_type(8)))  __bf16 bf16x8;
typedef __attribute__((ext_vector_type(8)))  float   f32x8;

static __device__ __forceinline__ f32x8 wmma_bf16(bf16x16 a, bf16x16 b, f32x8 c) {
  return __builtin_amdgcn_wmma_f32_16x16x32_bf16(false, a, false, b, (short)0, c,
                                                 false, false);
}

// ---------------------------------------------------------------------------
// Small utility kernels
// ---------------------------------------------------------------------------
__global__ void mask_atoms_k(int* xm, const int* __restrict__ idx, int nm) {
  int t = blockIdx.x * 256 + threadIdx.x;
  if (t >= nm * 7) return;
  int i = t / 7, c = t % 7;
  const int vals[7] = {118, 9, 11, 11, 9, 5, 5};  // ATOM_VOCABS[c]-1
  xm[(size_t)idx[i] * 7 + c] = vals[c];
}

__global__ void mask_bonds_k(int* bm, const int* __restrict__ idx, int nm) {
  int t = blockIdx.x * 256 + threadIdx.x;
  if (t >= nm * 3) return;
  int i = t / 3, c = t % 3;
  const int vals[3] = {4, 5, 1};  // BOND_VOCABS[c]-1
  bm[(size_t)idx[i] * 3 + c] = vals[c];
}

__global__ void mask_zero_f_k(float* v, const int* __restrict__ idx, int nm) {
  int t = blockIdx.x * 256 + threadIdx.x;
  if (t >= nm) return;
  v[idx[t]] = 0.0f;
}

__global__ void count_atoms_k(const int* __restrict__ batch, float* counts, int n) {
  int t = blockIdx.x * 256 + threadIdx.x;
  if (t >= n) return;
  atomicAdd(&counts[batch[t]], 1.0f);
}

__global__ void scan_k(const int* __restrict__ num, int* off, int G) {
  if (blockIdx.x == 0 && threadIdx.x == 0) {
    int s = 0;
    for (int g = 0; g < G; ++g) { off[g] = s; s += num[g]; }
    off[G] = s;
  }
}

__global__ void fill_batch_k(const int* __restrict__ off, int G, int* batch, int N) {
  int e = blockIdx.x * 256 + threadIdx.x;
  if (e >= N) return;
  int lo = 0, hi = G - 1;
  while (lo < hi) {
    int mid = (lo + hi + 1) >> 1;
    if (off[mid] <= e) lo = mid; else hi = mid - 1;
  }
  batch[e] = lo;
}

__global__ void gnorm_f_k(const float* __restrict__ c, float* o, int G) {
  int t = blockIdx.x * 256 + threadIdx.x;
  if (t >= G) return;
  o[t] = rsqrtf(c[t]);
}

__global__ void gnorm_i_k(const int* __restrict__ c, float* o, int G) {
  int t = blockIdx.x * 256 + threadIdx.x;
  if (t >= G) return;
  o[t] = rsqrtf((float)c[t]);
}

// ---------------------------------------------------------------------------
// Embedding gathers
// ---------------------------------------------------------------------------
__global__ void embed7_k(const int* __restrict__ xm,
                         const float* __restrict__ t0, const float* __restrict__ t1,
                         const float* __restrict__ t2, const float* __restrict__ t3,
                         const float* __restrict__ t4, const float* __restrict__ t5,
                         const float* __restrict__ t6,
                         float* __restrict__ out, int rows) {
  int t = blockIdx.x * 256 + threadIdx.x;
  int row = t >> 7;
  if (row >= rows) return;
  int j = t & 127;
  const int* xr = xm + (size_t)row * 7;
  float a = t0[(size_t)xr[0] * DD + j] + t1[(size_t)xr[1] * DD + j] +
            t2[(size_t)xr[2] * DD + j] + t3[(size_t)xr[3] * DD + j] +
            t4[(size_t)xr[4] * DD + j] + t5[(size_t)xr[5] * DD + j] +
            t6[(size_t)xr[6] * DD + j];
  out[(size_t)row * DD + j] = a;
}

__global__ void embed3_k(const int* __restrict__ xm,
                         const float* __restrict__ t0, const float* __restrict__ t1,
                         const float* __restrict__ t2,
                         float* __restrict__ out, int rows) {
  int t = blockIdx.x * 256 + threadIdx.x;
  int row = t >> 7;
  if (row >= rows) return;
  int j = t & 127;
  const int* xr = xm + (size_t)row * 3;
  float a = t0[(size_t)xr[0] * DD + j] + t1[(size_t)xr[1] * DD + j] +
            t2[(size_t)xr[2] * DD + j];
  out[(size_t)row * DD + j] = a;
}

// ---------------------------------------------------------------------------
// RBF: out[row,:] (+)= exp(-10*(v-c_k)^2) @ W[K,128] + b
// ---------------------------------------------------------------------------
__global__ void rbf_k(const float* __restrict__ vals, const float* __restrict__ W,
                      const float* __restrict__ b, float* __restrict__ out,
                      int K, float c0, int accumulate) {
  __shared__ float r[64];
  int row = blockIdx.x;
  int j = threadIdx.x;  // 128 threads
  float v = vals[row];
  if (j < K) {
    float d = v - (c0 + 0.1f * (float)j);
    r[j] = __expf(-10.0f * d * d);
  }
  __syncthreads();
  float acc = accumulate ? out[(size_t)row * DD + j] : 0.0f;
  acc += b[j];
  for (int k = 0; k < K; ++k) acc = fmaf(r[k], W[(size_t)k * DD + j], acc);
  out[(size_t)row * DD + j] = acc;
}

// ---------------------------------------------------------------------------
// Edge scatter: agg[dst[e]] += nodeH[src[e]] + edgeH[e]   (float4 per thread)
// ---------------------------------------------------------------------------
__global__ void scatter_k(const int* __restrict__ src, const int* __restrict__ dst,
                          const float* __restrict__ nodeH, const float* __restrict__ edgeH,
                          float* __restrict__ agg, int E) {
  int t = blockIdx.x * 256 + threadIdx.x;
  int e = t >> 5;
  if (e >= E) return;
  int c = (t & 31) * 4;
  int s = src[e], d = dst[e];
  const float4 a = *(const float4*)&nodeH[(size_t)s * DD + c];
  const float4 b = *(const float4*)&edgeH[(size_t)e * DD + c];
  float* p = &agg[(size_t)d * DD + c];
  atomicAdd(p + 0, a.x + b.x);
  atomicAdd(p + 1, a.y + b.y);
  atomicAdd(p + 2, a.z + b.z);
  atomicAdd(p + 3, a.w + b.w);
}

// ---------------------------------------------------------------------------
// Weight repack: f32 [K,N] row-major -> bf16 WMMA B-fragments.
// Fragment f = nt*(K/32)+kt ; lane holds 16 bf16 (A/B 16-bit VGPR layout):
//   n = nt*16 + (lane&15);  k = kt*32 + (i<8? i : i+8) + (lane>=16 ? 8 : 0)
// ---------------------------------------------------------------------------
__global__ void repack_w_k(const float* __restrict__ W, bf16* __restrict__ out,
                           int K, int N) {
  int lane = threadIdx.x;  // 32 threads
  int KT = K >> 5;
  int nt = blockIdx.x / KT;
  int kt = blockIdx.x % KT;
  int n = nt * 16 + (lane & 15);
  int hi8 = (lane >> 4) ? 8 : 0;
  bf16* o = out + ((size_t)blockIdx.x * 32 + lane) * 16;
#pragma unroll
  for (int i = 0; i < 16; ++i) {
    int koff = (i < 8 ? i : i + 8) + hi8;
    int k = kt * 32 + koff;
    o[i] = (bf16)W[(size_t)k * N + n];
  }
}

// ---------------------------------------------------------------------------
// Fused GIN block MLP on WMMA:
//   h = relu(X@W1+b1)@W2+b2 ; h = LN(h)*g+beta ; h *= gnorm[batch] ;
//   (relu) ; out = h + resid
// One wave = 16 rows. 4 waves / block (128 threads), 32KB LDS.
// ---------------------------------------------------------------------------
__launch_bounds__(128)
__global__ void mlp_block_k(const float* __restrict__ X, const float* __restrict__ resid,
                            const bf16* __restrict__ W1p, const float* __restrict__ b1,
                            const bf16* __restrict__ W2p, const float* __restrict__ b2,
                            const float* __restrict__ gma, const float* __restrict__ bta,
                            const int* __restrict__ batch, const float* __restrict__ gnorm,
                            float* __restrict__ out, int rows, int act) {
  __shared__ __align__(16) unsigned char lds[4][16 * HH * 2];  // 8KB / wave
  const int wave = threadIdx.x >> 5;
  const int lane = threadIdx.x & 31;
  const int rowBase = blockIdx.x * 64 + wave * 16;
  const int m = lane & 15;
  const int hi8 = (lane >> 4) ? 8 : 0;  // high half-wave handles K+8 / M+8
  int arow = rowBase + m;
  if (arow >= rows) arow = rows - 1;  // clamp loads; stores skipped later

  // ---- A fragments for GEMM1 (16x128 tile, f32 -> bf16, b128 loads) ----
  bf16x16 af[4];
  {
    const float* xr = X + (size_t)arow * DD + hi8;
#pragma unroll
    for (int kt = 0; kt < 4; ++kt) {
      const float* p = xr + kt * 32;
      const float4 f0 = *(const float4*)(p + 0);
      const float4 f1 = *(const float4*)(p + 4);
      const float4 f2 = *(const float4*)(p + 16);
      const float4 f3 = *(const float4*)(p + 20);
      af[kt][0] = (bf16)f0.x;  af[kt][1] = (bf16)f0.y;
      af[kt][2] = (bf16)f0.z;  af[kt][3] = (bf16)f0.w;
      af[kt][4] = (bf16)f1.x;  af[kt][5] = (bf16)f1.y;
      af[kt][6] = (bf16)f1.z;  af[kt][7] = (bf16)f1.w;
      af[kt][8]  = (bf16)f2.x; af[kt][9]  = (bf16)f2.y;
      af[kt][10] = (bf16)f2.z; af[kt][11] = (bf16)f2.w;
      af[kt][12] = (bf16)f3.x; af[kt][13] = (bf16)f3.y;
      af[kt][14] = (bf16)f3.z; af[kt][15] = (bf16)f3.w;
    }
  }

  // ---- GEMM1: h1[16,256] = relu(X@W1 + b1), staged bf16 in LDS ----
  bf16* h1 = (bf16*)lds[wave];
#pragma unroll 1
  for (int nt = 0; nt < 16; ++nt) {
    f32x8 acc = {};
#pragma unroll
    for (int kt = 0; kt < 4; ++kt) {
      bf16x16 bf = *(const bf16x16*)(W1p + (((size_t)nt * 4 + kt) * 32 + lane) * 16);
      acc = wmma_bf16(af[kt], bf, acc);
    }
    const int col = nt * 16 + m;
    const float bias = b1[col];
#pragma unroll
    for (int r = 0; r < 8; ++r) {
      float v = acc[r] + bias;
      v = v > 0.0f ? v : 0.0f;
      h1[(size_t)(hi8 + r) * HH + col] = (bf16)v;  // D layout: M = r (+8 for hi lanes)
    }
  }
  __syncthreads();

  // ---- re-fragment h1 (A fragments for GEMM2), 2x ds_load_b128 per frag ----
  bf16x16 a2[8];
  {
    const bf16* bp = h1 + (size_t)m * HH + hi8;
#pragma unroll
    for (int kt = 0; kt < 8; ++kt) {
      const bf16* p = bp + kt * 32;
      bf16x8 lo = *(const bf16x8*)(p + 0);
      bf16x8 hi = *(const bf16x8*)(p + 16);
      a2[kt] = __builtin_shufflevector(lo, hi, 0, 1, 2, 3, 4, 5, 6, 7,
                                       8, 9, 10, 11, 12, 13, 14, 15);
    }
  }
  __syncthreads();

  // ---- GEMM2: ho[16,128] = h1@W2 + b2 (LDS reused as f32) ----
  float* ho = (float*)lds[wave];
#pragma unroll 1
  for (int nt = 0; nt < 8; ++nt) {
    f32x8 acc = {};
#pragma unroll
    for (int kt = 0; kt < 8; ++kt) {
      bf16x16 bf = *(const bf16x16*)(W2p + (((size_t)nt * 8 + kt) * 32 + lane) * 16);
      acc = wmma_bf16(a2[kt], bf, acc);
    }
    const int col = nt * 16 + m;
    const float bias = b2[col];
#pragma unroll
    for (int r = 0; r < 8; ++r) ho[(size_t)(hi8 + r) * DD + col] = acc[r] + bias;
  }
  __syncthreads();

  // ---- epilogue: LayerNorm + GraphNorm + (ReLU) + residual ----
  const float4 g4  = *(const float4*)&gma[lane * 4];
  const float4 be4 = *(const float4*)&bta[lane * 4];
  for (int r = 0; r < 16; ++r) {
    int row = rowBase + r;
    if (row >= rows) break;  // wave-uniform
    float4 v = *(const float4*)&ho[(size_t)r * DD + lane * 4];
    float s = v.x + v.y + v.z + v.w;
#pragma unroll
    for (int o = 16; o; o >>= 1) s += __shfl_xor(s, o, 32);
    float mean = s * (1.0f / 128.0f);
    float dx = v.x - mean, dy = v.y - mean, dz = v.z - mean, dw = v.w - mean;
    float s2 = dx * dx + dy * dy + dz * dz + dw * dw;
#pragma unroll
    for (int o = 16; o; o >>= 1) s2 += __shfl_xor(s2, o, 32);
    float inv = rsqrtf(s2 * (1.0f / 128.0f) + 1e-5f);
    float gn = gnorm[batch[row]];
    float4 res = *(const float4*)&resid[(size_t)row * DD + lane * 4];
    float o0 = (dx * inv * g4.x + be4.x) * gn;
    float o1 = (dy * inv * g4.y + be4.y) * gn;
    float o2 = (dz * inv * g4.z + be4.z) * gn;
    float o3 = (dw * inv * g4.w + be4.w) * gn;
    if (act) {
      o0 = fmaxf(o0, 0.f); o1 = fmaxf(o1, 0.f);
      o2 = fmaxf(o2, 0.f); o3 = fmaxf(o3, 0.f);
    }
    float4 ov = make_float4(o0 + res.x, o1 + res.y, o2 + res.z, o3 + res.w);
    *(float4*)&out[(size_t)row * DD + lane * 4] = ov;
  }
}

// ---------------------------------------------------------------------------
// Graph readout
// ---------------------------------------------------------------------------
__global__ void graph_accum_k(const float* __restrict__ nodeH, const int* __restrict__ batch,
                              float* __restrict__ gsum, int rows) {
  int t = blockIdx.x * 256 + threadIdx.x;
  int row = t >> 5;
  if (row >= rows) return;
  int c = (t & 31) * 4;
  int g = batch[row];
  const float4 v = *(const float4*)&nodeH[(size_t)row * DD + c];
  float* p = &gsum[(size_t)g * DD + c];
  atomicAdd(p + 0, v.x);
  atomicAdd(p + 1, v.y);
  atomicAdd(p + 2, v.z);
  atomicAdd(p + 3, v.w);
}

__global__ void graph_scale_k(float* gsum, const float* __restrict__ counts, int G) {
  int t = blockIdx.x * 256 + threadIdx.x;
  int gi = t >> 7;
  if (gi >= G) return;
  int j = t & 127;
  gsum[(size_t)gi * DD + j] /= counts[gi];
}

// ---------------------------------------------------------------------------
// Host orchestration
// ---------------------------------------------------------------------------
extern "C" void kernel_launch(void* const* d_in, const int* in_sizes, int n_in,
                              void* d_out, int out_size, void* d_ws, size_t ws_size,
                              hipStream_t stream) {
  if (n_in < 111) return;  // expects 16 positional inputs + 95 param leaves

  auto cdiv = [](long long a, long long b) -> int { return (int)((a + b - 1) / b); };
  auto F = [&](int i) -> const float* { return (const float*)d_in[i]; };

  const int* AB = (const int*)d_in[0];
  const int* BA = (const int*)d_in[1];
  const int* AD = (const int*)d_in[2];
  const int* x_in = (const int*)d_in[3];
  const int* battr_in = (const int*)d_in[4];
  const float* blen_in = (const float*)d_in[5];
  const float* bang_in = (const float*)d_in[6];
  const float* dih_in = (const float*)d_in[7];
  const int* atom_batch = (const int*)d_in[8];
  const int* num_bonds = (const int*)d_in[9];
  const int* num_angles = (const int*)d_in[10];
  const int* m_atom = (const int*)d_in[12];
  const int* m_bond = (const int*)d_in[13];
  const int* m_angle = (const int*)d_in[14];
  const int* m_dih = (const int*)d_in[15];

  const int NB = in_sizes[0] / 2;    // bonds (= AB edges)
  const int NANG = in_sizes[1] / 2;  // angles (= BA edges)
  const int ND = in_sizes[2] / 2;    // dihedrals (= AD edges)
  const int NATOM = in_sizes[3] / 7;
  const int G = in_sizes[9];
  const int nmA = in_sizes[12], nmB = in_sizes[13], nmAn = in_sizes[14], nmD = in_sizes[15];

  // params (jax pytree, sorted dict keys, appended after the 16 inputs)
  const float* angInitW = F(16);
  const float* angInitB = F(17);
  const float* atomEmb[7] = {F(18), F(19), F(20), F(21), F(22), F(23), F(24)};
  const float* bondEmbI[3] = {F(25), F(26), F(27)};
  const float* bondInitW = F(28);
  const float* bondInitB = F(29);

  // ---- carve workspace ----
  char* wsb = (char*)d_ws;
  size_t off = 0;
  auto alloc = [&](size_t bytes) -> void* {
    void* p = wsb + off;
    off = (off + bytes + 255) & ~(size_t)255;
    return p;
  };
  float* nodeA = (float*)alloc((size_t)NATOM * DD * 4);
  float* nodeB = (float*)alloc((size_t)NATOM * DD * 4);
  float* bondH = (float*)alloc((size_t)NB * DD * 4);
  float* curEdge = (float*)alloc((size_t)NB * DD * 4);
  float* angleH = (float*)alloc((size_t)NANG * DD * 4);
  float* curAngle = (float*)alloc((size_t)NANG * DD * 4);
  float* agg = (float*)alloc((size_t)NANG * DD * 4);  // max rows = angles
  int* xm = (int*)alloc((size_t)NATOM * 7 * 4);
  int* battr = (int*)alloc((size_t)NB * 3 * 4);
  float* blen = (float*)alloc((size_t)NB * 4);
  float* bang = (float*)alloc((size_t)NANG * 4);
  float* dihm = (float*)alloc((size_t)ND * 4);
  int* bbatch = (int*)alloc((size_t)NB * 4);
  int* abatch = (int*)alloc((size_t)NANG * 4);
  float* counts = (float*)alloc((size_t)G * 4);
  float* gnA = (float*)alloc((size_t)G * 4);
  float* gnB = (float*)alloc((size_t)G * 4);
  float* gnAn = (float*)alloc((size_t)G * 4);
  int* boff = (int*)alloc((size_t)(G + 1) * 4);
  int* aoff = (int*)alloc((size_t)(G + 1) * 4);
  bf16* wpack = (bf16*)alloc((size_t)9 * 65536 * 2);  // 9 blocks x (W1p+W2p)

  float* outNode = (float*)d_out;
  float* outBond = outNode + (size_t)NATOM * DD;
  float* outAngle = outBond + (size_t)NB * DD;
  float* outDih = outAngle + (size_t)NANG * DD;
  float* outGraph = outDih + (size_t)ND * DD;

  const float PI = 3.14159265358979323846f;

  // ---- masked input copies ----
  hipMemcpyAsync(xm, x_in, (size_t)NATOM * 7 * 4, hipMemcpyDeviceToDevice, stream);
  hipMemcpyAsync(battr, battr_in, (size_t)NB * 3 * 4, hipMemcpyDeviceToDevice, stream);
  hipMemcpyAsync(blen, blen_in, (size_t)NB * 4, hipMemcpyDeviceToDevice, stream);
  hipMemcpyAsync(bang, bang_in, (size_t)NANG * 4, hipMemcpyDeviceToDevice, stream);
  hipMemcpyAsync(dihm, dih_in, (size_t)ND * 4, hipMemcpyDeviceToDevice, stream);
  mask_atoms_k<<<cdiv((long long)nmA * 7, 256), 256, 0, stream>>>(xm, m_atom, nmA);
  mask_bonds_k<<<cdiv((long long)nmB * 3, 256), 256, 0, stream>>>(battr, m_bond, nmB);
  mask_zero_f_k<<<cdiv(nmB, 256), 256, 0, stream>>>(blen, m_bond, nmB);
  mask_zero_f_k<<<cdiv(nmAn, 256), 256, 0, stream>>>(bang, m_angle, nmAn);
  mask_zero_f_k<<<cdiv(nmD, 256), 256, 0, stream>>>(dihm, m_dih, nmD);

  // ---- batch vectors, counts, graph-norm factors ----
  hipMemsetAsync(counts, 0, (size_t)G * 4, stream);
  count_atoms_k<<<cdiv(NATOM, 256), 256, 0, stream>>>(atom_batch, counts, NATOM);
  scan_k<<<1, 1, 0, stream>>>(num_bonds, boff, G);
  scan_k<<<1, 1, 0, stream>>>(num_angles, aoff, G);
  fill_batch_k<<<cdiv(NB, 256), 256, 0, stream>>>(boff, G, bbatch, NB);
  fill_batch_k<<<cdiv(NANG, 256), 256, 0, stream>>>(aoff, G, abatch, NANG);
  gnorm_f_k<<<cdiv(G, 256), 256, 0, stream>>>(counts, gnA, G);
  gnorm_i_k<<<cdiv(G, 256), 256, 0, stream>>>(num_bonds, gnB, G);
  gnorm_i_k<<<cdiv(G, 256), 256, 0, stream>>>(num_angles, gnAn, G);

  // ---- repack all block weights to bf16 WMMA fragments ----
  // layer leaf base = 30 + 27*l; block leaf order (sorted): W1,W2,b1,b2,beta,g
  for (int l = 0; l < 3; ++l) {
    int Lb = 30 + 27 * l;
    const int blkOff[3] = {0, 14, 6};  // ab, ba, ad leaf offsets within layer
    for (int b = 0; b < 3; ++b) {
      const float* W1 = F(Lb + blkOff[b] + 0);
      const float* W2 = F(Lb + blkOff[b] + 1);
      bf16* w1p = wpack + (size_t)(l * 3 + b) * 65536;
      bf16* w2p = w1p + 32768;
      repack_w_k<<<64, 32, 0, stream>>>(W1, w1p, 128, 256);
      repack_w_k<<<64, 32, 0, stream>>>(W2, w2p, 256, 128);
    }
  }

  // ---- initial embeddings ----
  embed7_k<<<cdiv((long long)NATOM * DD, 256), 256, 0, stream>>>(
      xm, atomEmb[0], atomEmb[1], atomEmb[2], atomEmb[3], atomEmb[4], atomEmb[5],
      atomEmb[6], nodeA, NATOM);
  embed3_k<<<cdiv((long long)NB * DD, 256), 256, 0, stream>>>(
      battr, bondEmbI[0], bondEmbI[1], bondEmbI[2], bondH, NB);
  rbf_k<<<NB, 128, 0, stream>>>(blen, bondInitW, bondInitB, bondH, 20, 0.0f, 1);
  rbf_k<<<NANG, 128, 0, stream>>>(bang, angInitW, angInitB, angleH, 32, 0.0f, 0);

  float* nodeCur = nodeA;
  float* nodeNxt = nodeB;

  for (int l = 0; l < 3; ++l) {
    int act = (l != 2) ? 1 : 0;
    int Lb = 30 + 27 * l;
    const float* abW[6], *baW[6], *adW[6];
    for (int i = 0; i < 6; ++i) abW[i] = F(Lb + 0 + i);
    for (int i = 0; i < 6; ++i) adW[i] = F(Lb + 6 + i);
    const float* angW = F(Lb + 12), *angB = F(Lb + 13);
    for (int i = 0; i < 6; ++i) baW[i] = F(Lb + 14 + i);
    const float* bE0 = F(Lb + 20), *bE1 = F(Lb + 21), *bE2 = F(Lb + 22);
    const float* brW = F(Lb + 23), *brB = F(Lb + 24);
    const float* drW = F(Lb + 25), *drB = F(Lb + 26);
    bf16* abW1p = wpack + (size_t)(l * 3 + 0) * 65536, *abW2p = abW1p + 32768;
    bf16* baW1p = wpack + (size_t)(l * 3 + 1) * 65536, *baW2p = baW1p + 32768;
    bf16* adW1p = wpack + (size_t)(l * 3 + 2) * 65536, *adW2p = adW1p + 32768;

    // --- atom block (AB): agg over bonds -> atoms ---
    hipMemsetAsync(agg, 0, (size_t)NATOM * DD * 4, stream);
    scatter_k<<<cdiv((long long)NB * 32, 256), 256, 0, stream>>>(
        AB, AB + NB, nodeCur, bondH, agg, NB);
    mlp_block_k<<<cdiv(NATOM, 64), 128, 0, stream>>>(
        agg, nodeCur, abW1p, abW[2], abW2p, abW[3], abW[5], abW[4],
        atom_batch, gnA, nodeNxt, NATOM, act);

    // --- layer-fresh bond features ---
    embed3_k<<<cdiv((long long)NB * DD, 256), 256, 0, stream>>>(
        battr, bE0, bE1, bE2, curEdge, NB);
    rbf_k<<<NB, 128, 0, stream>>>(blen, brW, brB, curEdge, 20, 0.0f, 1);

    // --- bond block (BA): agg over angles -> bonds (edge feats = old angleH) ---
    hipMemsetAsync(agg, 0, (size_t)NB * DD * 4, stream);
    scatter_k<<<cdiv((long long)NANG * 32, 256), 256, 0, stream>>>(
        BA, BA + NANG, curEdge, angleH, agg, NANG);
    mlp_block_k<<<cdiv(NB, 64), 128, 0, stream>>>(
        agg, curEdge, baW1p, baW[2], baW2p, baW[3], baW[5], baW[4],
        bbatch, gnB, bondH, NB, act);

    // --- layer-fresh angle + dihedral features (dih straight into d_out) ---
    rbf_k<<<NANG, 128, 0, stream>>>(bang, angW, angB, curAngle, 32, 0.0f, 0);
    rbf_k<<<ND, 128, 0, stream>>>(dihm, drW, drB, outDih, 63, -PI, 0);

    // --- angle block (AD): agg over dihedrals -> angles ---
    hipMemsetAsync(agg, 0, (size_t)NANG * DD * 4, stream);
    scatter_k<<<cdiv((long long)ND * 32, 256), 256, 0, stream>>>(
        AD, AD + ND, curAngle, outDih, agg, ND);
    mlp_block_k<<<cdiv(NANG, 64), 128, 0, stream>>>(
        agg, curAngle, adW1p, adW[2], adW2p, adW[3], adW[5], adW[4],
        abatch, gnAn, angleH, NANG, act);

    float* tmp = nodeCur; nodeCur = nodeNxt; nodeNxt = tmp;
  }

  // ---- outputs ----
  hipMemcpyAsync(outNode, nodeCur, (size_t)NATOM * DD * 4, hipMemcpyDeviceToDevice, stream);
  hipMemcpyAsync(outBond, bondH, (size_t)NB * DD * 4, hipMemcpyDeviceToDevice, stream);
  hipMemcpyAsync(outAngle, angleH, (size_t)NANG * DD * 4, hipMemcpyDeviceToDevice, stream);
  hipMemsetAsync(outGraph, 0, (size_t)G * DD * 4, stream);
  graph_accum_k<<<cdiv((long long)NATOM * 32, 256), 256, 0, stream>>>(
      nodeCur, atom_batch, outGraph, NATOM);
  graph_scale_k<<<cdiv((long long)G * DD, 256), 256, 0, stream>>>(outGraph, counts, G);

  (void)out_size; (void)ws_size;
}
